// MetalWallQEQ_83262236000860
// MI455X (gfx1250) — compile-verified, matrix operand order checked by reading, not run on hardware
//
#include <hip/hip_runtime.h>

// ---------------- problem constants ----------------
#define NATOM 4096
#define NMET  1024
#define NKV   9260      // (2*10+1)^3 - 1 k-vectors
#define KPAD  9280      // K padded to multiple of 32 (WMMA K-step)
#define LDK   KPAD
#define NA    1025      // bordered system size
#define NCOL  1026      // augmented columns (incl. chi)
#define LDM   1056      // row stride of augmented matrix (floats)

// GEMM tiling
#define KC    32        // K-chunk staged in LDS per iteration
#define LROW  40        // padded LDS row stride in elements (80B -> conflict-free b128 reads)
#define BUFE  (64 * LROW) // elements per LDS panel (64 rows)

typedef __attribute__((ext_vector_type(16))) __bf16 v16bf;
typedef __attribute__((ext_vector_type(8)))  __bf16 v8bf;
typedef __attribute__((ext_vector_type(8)))  float  v8f;

// ---------------- bf16 helpers (RNE) ----------------
static __device__ inline __bf16 f2bf(float f) {
  unsigned u = __float_as_uint(f);
  u += 0x7FFFu + ((u >> 16) & 1u);
  unsigned short h = (unsigned short)(u >> 16);
  return __builtin_bit_cast(__bf16, h);
}
static __device__ inline float bf2f(__bf16 b) {
  unsigned short h = __builtin_bit_cast(unsigned short, b);
  return __uint_as_float(((unsigned)h) << 16);
}

// Per-lane A/B fragment for V_WMMA_F32_16X16X32_BF16:
// row = lane&15, K-chunks {base..base+7} and {base+16..base+23}, base = 8*(lane>>4)
struct FragPair { v8bf a, b; };
static __device__ inline v16bf load_frag(const __bf16* p) {
  FragPair t;
  t.a = *reinterpret_cast<const v8bf*>(p);
  t.b = *reinterpret_cast<const v8bf*>(p + 16);
  return __builtin_bit_cast(v16bf, t);
}

// ---------------- CDNA5 async global->LDS copy (ASYNCcnt path) ----------------
static __device__ inline void async_ld_b128(unsigned ldsOff, const void* gptr) {
  // VDST = VGPR with LDS byte offset, VADDR = 64-bit global address, GV mode
  asm volatile("global_load_async_to_lds_b128 %0, %1, off" :: "v"(ldsOff), "v"(gptr) : "memory");
}
static __device__ inline void wait_async() {
  asm volatile("s_wait_asynccnt 0x0" ::: "memory");
}

// ---------------- K1: k-vector table ----------------
__global__ void k_build_ktab(const float* __restrict__ cell, float4* __restrict__ ktab) {
  int j = blockIdx.x * blockDim.x + threadIdx.x;
  if (j >= KPAD) return;
  if (j >= NKV) { ktab[j] = make_float4(0.f, 0.f, 0.f, 0.f); return; }
  float c00=cell[0],c01=cell[1],c02=cell[2];
  float c10=cell[3],c11=cell[4],c12=cell[5];
  float c20=cell[6],c21=cell[7],c22=cell[8];
  float det = c00*(c11*c22-c12*c21) - c01*(c10*c22-c12*c20) + c02*(c10*c21-c11*c20);
  float vol = fabsf(det);
  float id  = 1.0f / det;
  float i00 =  (c11*c22 - c12*c21)*id, i01 = -(c01*c22 - c02*c21)*id, i02 =  (c01*c12 - c02*c11)*id;
  float i10 = -(c10*c22 - c12*c20)*id, i11 =  (c00*c22 - c02*c20)*id, i12 = -(c00*c12 - c02*c10)*id;
  float i20 =  (c10*c21 - c11*c20)*id, i21 = -(c00*c21 - c01*c20)*id, i22 =  (c00*c11 - c01*c10)*id;
  int g = (j < 4630) ? j : j + 1;                 // skip the (0,0,0) flat index
  float ia = (float)(g / 441 - 10);
  float ib = (float)((g / 21) % 21 - 10);
  float ic = (float)(g % 21 - 10);
  const float TWO_PI = 6.283185307179586f;
  float kx = TWO_PI * (i00*ia + i01*ib + i02*ic);
  float ky = TWO_PI * (i10*ia + i11*ib + i12*ic);
  float kz = TWO_PI * (i20*ia + i21*ib + i22*ic);
  float k2 = kx*kx + ky*ky + kz*kz;
  const float SIG = 1.0f / 1.805132f;
  float kfac = (12.566370614359172f / vol) * __expf(-0.5f * SIG * SIG * k2) / k2; // 4pi/V folded in
  ktab[j] = make_float4(kx, ky, kz, sqrtf(kfac));
}

// ---------------- K2: weighted cos/sin, bf16 hi/lo split ----------------
__global__ void k_build_cs(const float* __restrict__ pos, const float4* __restrict__ ktab,
                           __bf16* __restrict__ Ch, __bf16* __restrict__ Cl,
                           __bf16* __restrict__ Sh, __bf16* __restrict__ Sl) {
  int k = blockIdx.x * blockDim.x + threadIdx.x;
  int i = blockIdx.y;                     // metal atom
  if (k >= KPAD) return;
  float4 kt = ktab[k];
  float dot = pos[3*i]*kt.x + pos[3*i+1]*kt.y + pos[3*i+2]*kt.z;
  float s, c;
  __sincosf(dot, &s, &c);
  float cw = c * kt.w, sw = s * kt.w;     // fold sqrt(kfac')
  size_t idx = (size_t)i * LDK + k;
  __bf16 ch = f2bf(cw); Ch[idx] = ch; Cl[idx] = f2bf(cw - bf2f(ch));
  __bf16 sh = f2bf(sw); Sh[idx] = sh; Sl[idx] = f2bf(sw - bf2f(sh));
}

// ---------------- K3: Gram matrix A = Cw Cw^T + Sw Sw^T ----------------
// 8 waves / WG, 64x64 tile, LDS-staged panels via async copies, bf16x3 WMMA.
__global__ __launch_bounds__(256) void k_gram_wmma(const __bf16* __restrict__ Ch, const __bf16* __restrict__ Cl,
                                                   const __bf16* __restrict__ Sh, const __bf16* __restrict__ Sl,
                                                   float* __restrict__ Mm) {
  __shared__ __bf16 smem[8 * BUFE];       // 8 panels x 64 rows x 40 elems = 40 KB
  const int tid  = threadIdx.x;
  const int lane = tid & 31;
  const int wave = tid >> 5;              // 0..7
  const int wm   = wave >> 1;             // 0..3: 16-row M sub-tile
  const int wn   = wave & 1;              // 0..1: 32-col N sub-tile
  const int tm   = blockIdx.x * 64;
  const int tn   = blockIdx.y * 64;

  const __bf16* arrs[4] = {Ch, Cl, Sh, Sl};
  const int srow  = tid >> 2;             // 0..63: row this thread copies
  const int kpart = tid & 3;              // 0..3: 16B sub-chunk of the 64B row
  const unsigned ldsBase = (unsigned)(size_t)(const void*)&smem[0]; // flat-LDS low 32 bits == LDS offset

  const int arow  = wm * 16 + (lane & 15);
  const int brow0 = wn * 32 + (lane & 15);
  const int koff  = (lane >> 4) << 3;

  v8f acc0 = {}, acc1 = {};
  for (int k0 = 0; k0 < LDK; k0 += KC) {
    // stage A-panels (rows tm..tm+63) into LDS buffers 0..3 and B-panels (tn..) into 4..7
    #pragma unroll
    for (int r = 0; r < 8; ++r) {
      const __bf16* g = arrs[r & 3] +
          (size_t)(((r < 4) ? tm : tn) + srow) * LDK + (size_t)(k0 + kpart * 8);
      unsigned dst = ldsBase + (unsigned)(r * BUFE + srow * LROW + kpart * 8) * 2u;
      async_ld_b128(dst, (const void*)g);
    }
    wait_async();
    __syncthreads();

    v16bf aCh = load_frag(&smem[0*BUFE + arow*LROW + koff]);
    v16bf aCl = load_frag(&smem[1*BUFE + arow*LROW + koff]);
    v16bf aSh = load_frag(&smem[2*BUFE + arow*LROW + koff]);
    v16bf aSl = load_frag(&smem[3*BUFE + arow*LROW + koff]);
    v16bf b0Ch = load_frag(&smem[4*BUFE + brow0*LROW + koff]);
    v16bf b0Cl = load_frag(&smem[5*BUFE + brow0*LROW + koff]);
    v16bf b0Sh = load_frag(&smem[6*BUFE + brow0*LROW + koff]);
    v16bf b0Sl = load_frag(&smem[7*BUFE + brow0*LROW + koff]);
    v16bf b1Ch = load_frag(&smem[4*BUFE + (brow0+16)*LROW + koff]);
    v16bf b1Cl = load_frag(&smem[5*BUFE + (brow0+16)*LROW + koff]);
    v16bf b1Sh = load_frag(&smem[6*BUFE + (brow0+16)*LROW + koff]);
    v16bf b1Sl = load_frag(&smem[7*BUFE + (brow0+16)*LROW + koff]);

    // bf16x3 split: hi*hi + hi*lo + lo*hi, fp32 accumulate; A-frags reused for both N sub-tiles
    acc0 = __builtin_amdgcn_wmma_f32_16x16x32_bf16(false, aCh, false, b0Ch, (short)0, acc0, false, false);
    acc0 = __builtin_amdgcn_wmma_f32_16x16x32_bf16(false, aCh, false, b0Cl, (short)0, acc0, false, false);
    acc0 = __builtin_amdgcn_wmma_f32_16x16x32_bf16(false, aCl, false, b0Ch, (short)0, acc0, false, false);
    acc0 = __builtin_amdgcn_wmma_f32_16x16x32_bf16(false, aSh, false, b0Sh, (short)0, acc0, false, false);
    acc0 = __builtin_amdgcn_wmma_f32_16x16x32_bf16(false, aSh, false, b0Sl, (short)0, acc0, false, false);
    acc0 = __builtin_amdgcn_wmma_f32_16x16x32_bf16(false, aSl, false, b0Sh, (short)0, acc0, false, false);
    acc1 = __builtin_amdgcn_wmma_f32_16x16x32_bf16(false, aCh, false, b1Ch, (short)0, acc1, false, false);
    acc1 = __builtin_amdgcn_wmma_f32_16x16x32_bf16(false, aCh, false, b1Cl, (short)0, acc1, false, false);
    acc1 = __builtin_amdgcn_wmma_f32_16x16x32_bf16(false, aCl, false, b1Ch, (short)0, acc1, false, false);
    acc1 = __builtin_amdgcn_wmma_f32_16x16x32_bf16(false, aSh, false, b1Sh, (short)0, acc1, false, false);
    acc1 = __builtin_amdgcn_wmma_f32_16x16x32_bf16(false, aSh, false, b1Sl, (short)0, acc1, false, false);
    acc1 = __builtin_amdgcn_wmma_f32_16x16x32_bf16(false, aSl, false, b1Sh, (short)0, acc1, false, false);

    __syncthreads();   // LDS consumed; safe to overwrite next chunk
  }

  // C/D layout: VGPR r -> M = r (+8 for upper half-lanes), N = lane&15
  const int half = lane >> 4;
  const int c0   = tn + wn * 32 + (lane & 15);
  #pragma unroll
  for (int r = 0; r < 8; ++r) {
    int row = tm + wm * 16 + r + half * 8;
    Mm[(size_t)row * LDM + c0]      = acc0[r];
    Mm[(size_t)row * LDM + c0 + 16] = acc1[r];
  }
}

// ---------------- K4: structure factors over non-metal atoms ----------------
__global__ __launch_bounds__(256) void k_sf(const float* __restrict__ pos, const float* __restrict__ q,
                                            const float4* __restrict__ ktab,
                                            float* __restrict__ uc, float* __restrict__ us) {
  __shared__ float rc[256], rs[256];
  int k = blockIdx.x;
  float4 kt = ktab[k];
  float sc = 0.f, ss = 0.f;
  for (int i = NMET + threadIdx.x; i < NATOM; i += 256) {
    float dot = pos[3*i]*kt.x + pos[3*i+1]*kt.y + pos[3*i+2]*kt.z;
    float s, c; __sincosf(dot, &s, &c);
    float qi = q[i];
    sc += c * qi; ss += s * qi;
  }
  rc[threadIdx.x] = sc; rs[threadIdx.x] = ss;
  __syncthreads();
  for (int off = 128; off > 0; off >>= 1) {
    if (threadIdx.x < off) { rc[threadIdx.x] += rc[threadIdx.x + off]; rs[threadIdx.x] += rs[threadIdx.x + off]; }
    __syncthreads();
  }
  if (threadIdx.x == 0) {
    float kf = kt.w * kt.w;                    // kfac' = (sqrt(kfac'))^2
    uc[k] = kf * rc[0]; us[k] = kf * rs[0];
  }
}

// ---------------- K5: chi = -f[:NMET], written into augmented column ----------------
__global__ __launch_bounds__(256) void k_chi(const float* __restrict__ pos, const float4* __restrict__ ktab,
                                             const float* __restrict__ uc, const float* __restrict__ us,
                                             float* __restrict__ Mm) {
  __shared__ float red[256];
  int i = blockIdx.x;                          // metal atom
  float px = pos[3*i], py = pos[3*i+1], pz = pos[3*i+2];
  float acc = 0.f;
  for (int k = threadIdx.x; k < NKV; k += 256) {
    float4 kt = ktab[k];
    float dot = px*kt.x + py*kt.y + pz*kt.z;
    float s, c; __sincosf(dot, &s, &c);
    acc += c * uc[k] + s * us[k];
  }
  red[threadIdx.x] = acc;
  __syncthreads();
  for (int off = 128; off > 0; off >>= 1) {
    if (threadIdx.x < off) red[threadIdx.x] += red[threadIdx.x + off];
    __syncthreads();
  }
  if (threadIdx.x == 0) Mm[(size_t)i * LDM + NA] = -red[0];   // B = -f (SCALING = 1)
}

// ---------------- K6: borders of bordered system ----------------
__global__ void k_border(float* __restrict__ Mm) {
  int t = blockIdx.x * blockDim.x + threadIdx.x;
  if (t < NMET) {
    Mm[(size_t)t * LDM + NMET]    = 1.0f;      // last column of coeffs
    Mm[(size_t)NMET * LDM + t]    = 1.0f;      // last row of coeffs
  }
  if (t == 0) {
    Mm[(size_t)NMET * LDM + NMET] = 0.0f;      // corner
    Mm[(size_t)NMET * LDM + NA]   = 0.0f;      // chi[NMET]
  }
}

// ---------------- K7: Gauss-Jordan solve (single WG, pivot row in LDS) ----------------
__global__ __launch_bounds__(1024) void k_solve(float* __restrict__ Mm) {
  __shared__ float prow[NCOL];
  __shared__ float pinv_s;
  const int tid = threadIdx.x;
  for (int p = 0; p < NA; ++p) {
    if (tid == 0) pinv_s = 1.0f / Mm[(size_t)p * LDM + p];
    __syncthreads();
    float pinv = pinv_s;
    for (int j = tid; j < NCOL; j += 1024) {
      float v = Mm[(size_t)p * LDM + j] * pinv;
      prow[j] = v;
      Mm[(size_t)p * LDM + j] = v;
    }
    __syncthreads();
    for (int r = tid; r < NA; r += 1024) {
      if (r == p) continue;
      float f = Mm[(size_t)r * LDM + p];
      if (f != 0.0f) {
        for (int j = p; j < NCOL; ++j)
          Mm[(size_t)r * LDM + j] -= f * prow[j];
      }
    }
    __syncthreads();
  }
}

// ---------------- K8: scatter solution into output charges ----------------
__global__ void k_out(const float* __restrict__ q, const float* __restrict__ Mm, float* __restrict__ out) {
  int i = blockIdx.x * blockDim.x + threadIdx.x;
  if (i >= NATOM) return;
  out[i] = (i < NMET) ? Mm[(size_t)i * LDM + NA] : q[i];
}

// ---------------- launch ----------------
extern "C" void kernel_launch(void* const* d_in, const int* in_sizes, int n_in,
                              void* d_out, int out_size, void* d_ws, size_t ws_size,
                              hipStream_t stream) {
  (void)in_sizes; (void)n_in; (void)out_size; (void)ws_size;
  const float* pos  = (const float*)d_in[0];   // (4096,3)
  const float* q    = (const float*)d_in[1];   // (4096,1)
  const float* cell = (const float*)d_in[2];   // (1,3,3)
  float* out = (float*)d_out;

  char* ws = (char*)d_ws;
  float4* ktab = (float4*)(ws);                         // 9280*16  = 148,480 B
  float*  uc   = (float*)(ws + 0x40000);                // 9280*4
  float*  us   = (float*)(ws + 0x50000);                // 9280*4
  float*  Mm   = (float*)(ws + 0x60000);                // 1025*1056*4 = 4.33 MB
  __bf16* Ch   = (__bf16*)(ws + 0x500000);              // 4 x 1024*9280*2 = 76 MB
  __bf16* Cl   = Ch + (size_t)NMET * LDK;
  __bf16* Sh   = Cl + (size_t)NMET * LDK;
  __bf16* Sl   = Sh + (size_t)NMET * LDK;

  k_build_ktab<<<(KPAD + 255) / 256, 256, 0, stream>>>(cell, ktab);
  k_build_cs<<<dim3((KPAD + 255) / 256, NMET), 256, 0, stream>>>(pos, ktab, Ch, Cl, Sh, Sl);
  k_gram_wmma<<<dim3(NMET / 64, NMET / 64), 256, 0, stream>>>(Ch, Cl, Sh, Sl, Mm);
  k_sf<<<NKV, 256, 0, stream>>>(pos, q, ktab, uc, us);
  k_chi<<<NMET, 256, 0, stream>>>(pos, ktab, uc, us, Mm);
  k_border<<<(NMET + 255) / 256, 256, 0, stream>>>(Mm);
  k_solve<<<1, 1024, 0, stream>>>(Mm);
  k_out<<<(NATOM + 255) / 256, 256, 0, stream>>>(q, Mm, out);
}
